// HWxHW_Attention_33990371181525
// MI455X (gfx1250) — compile-verified, matrix operand order checked by reading, not run on hardware
//
#include <hip/hip_runtime.h>
#include <hip/hip_bf16.h>
#include <stdint.h>

// ---------------------------------------------------------------------------
// Problem constants (B=2, DIM=192, H=W=64, HEADS=4, c=48 padded to 64)
// ---------------------------------------------------------------------------
#define NB    2
#define DIM   192
#define HTOK  64
#define NTOK  4096          // H*W
#define HEADS 4
#define CH    48            // per-head channels
#define CP    64            // padded per-head channels (2 x K=32 WMMA steps)

typedef __attribute__((ext_vector_type(16))) _Float16 v16h;
typedef __attribute__((ext_vector_type(8)))  _Float16 v8h;
typedef __attribute__((ext_vector_type(8)))  float    v8f;
typedef __attribute__((ext_vector_type(4)))  int      v4i;

// ---------------------------------------------------------------------------
// CDNA5 async global->LDS copies (ASYNCcnt-tracked). Guarded so the file
// still builds on toolchains that lack the builtins (sync fallback).
// Builtin signature (from hipcc diagnostic): (int4 __device__*, int4
// __shared__*, imm offset, imm cpol).
// ---------------------------------------------------------------------------
#if __has_builtin(__builtin_amdgcn_global_load_async_to_lds_b128) && \
    __has_builtin(__builtin_amdgcn_s_wait_asynccnt)
#define HAVE_ASYNC_LDS 1
#else
#define HAVE_ASYNC_LDS 0
#endif

typedef __attribute__((address_space(1))) v4i as1_v4i;
typedef __attribute__((address_space(3))) v4i as3_v4i;

__device__ __forceinline__ void cp16_async(const _Float16* g, _Float16* l) {
#if HAVE_ASYNC_LDS
    __builtin_amdgcn_global_load_async_to_lds_b128(
        (as1_v4i*)(uintptr_t)g, (as3_v4i*)(uintptr_t)l, 0, 0);
#else
    *(v8h*)l = *(const v8h*)g;
#endif
}

template <int N>
__device__ __forceinline__ void wait_async() {
#if HAVE_ASYNC_LDS
    __builtin_amdgcn_s_wait_asynccnt(N);
#endif
}

// D = A(16x32 f16) * B(32x16 f16) + C(16x16 f32)
__device__ __forceinline__ v8f wmma_f32_f16(v16h a, v16h b, v8f c) {
    return __builtin_amdgcn_wmma_f32_16x16x32_f16(
        /*neg_a=*/false, a, /*neg_b=*/false, b,
        /*c_mod=*/(short)0, c, /*reuse_a=*/false, /*reuse_b=*/false);
}

// A-fragment (16x32, f16) from row-major f16 [16+ rows, ld cols] at p.
// ISA layout: lane l -> row m=l&15, k-base kb=(l>>4)*8,
//   elements 0..7  = cols kb..kb+7
//   elements 8..15 = cols kb+16..kb+23
// Also valid as B-fragment (32x16) when the source is stored n-major
// (i.e. src[n][k] row-major), which is how all our B operands are laid out.
__device__ __forceinline__ v16h frag_ld_h(const _Float16* p, int ld) {
    int l = threadIdx.x & 31;
    const _Float16* r = p + (l & 15) * ld + ((l >> 4) << 3);
    v8h lo = *(const v8h*)r;
    v8h hi = *(const v8h*)(r + 16);
    v16h o;
#pragma unroll
    for (int e = 0; e < 8; ++e) { o[e] = lo[e]; o[e + 8] = hi[e]; }
    return o;
}

// Same fragment, sourced from row-major f32 (weights), converted on the fly.
__device__ __forceinline__ v16h frag_ld_f32(const float* p, int ld) {
    int l = threadIdx.x & 31;
    const float* r = p + (l & 15) * ld + ((l >> 4) << 3);
    v16h o;
#pragma unroll
    for (int e = 0; e < 8; ++e) {
        o[e]     = (_Float16)r[e];
        o[e + 8] = (_Float16)r[e + 16];
    }
    return o;
}

// Store D (16x16 f32) to row-major f32 at p with leading dim ld.
// Lane l -> col n=l&15, rows (l>=16 ? 8 : 0)+r.
__device__ __forceinline__ void frag_st_f32(float* p, int ld, v8f d) {
    int l = threadIdx.x & 31;
    int n = l & 15, mo = (l >> 4) << 3;
#pragma unroll
    for (int r = 0; r < 8; ++r) p[(size_t)(mo + r) * ld + n] = d[r];
}

// ---------------------------------------------------------------------------
// Kernel 1: qkv 1x1 conv == GEMM  W[576x192] * X[192xNTOK]  (per batch)
// Block: 256 thr (8 waves). Block tile: 16 out-ch x 128 tokens.
// X tile is transposed into LDS (token-major) so B-fragments are contiguous.
// (stays synchronous: the staging does f32->f16 conversion in flight)
// ---------------------------------------------------------------------------
__global__ __launch_bounds__(256) void qkv1x1_kernel(
        const float* __restrict__ x, const float* __restrict__ w,
        float* __restrict__ outq) {
    __shared__ __align__(16) _Float16 Blds[128][32];
    const int tb = blockIdx.x;           // token block (128 tokens)
    const int ot = blockIdx.y;           // out-channel tile (16 rows)
    const int b  = blockIdx.z;
    const int wv = threadIdx.x >> 5;
    const float* xb = x + (size_t)b * DIM * NTOK;
    v8f acc = {};
#pragma unroll 1
    for (int kc = 0; kc < DIM / 32; ++kc) {
        __syncthreads();                 // protect Blds across iterations
        for (int i = threadIdx.x; i < 32 * 128; i += 256) {
            int cc = i >> 7, tk = i & 127;
            Blds[tk][cc] =
                (_Float16)xb[(size_t)(kc * 32 + cc) * NTOK + tb * 128 + tk];
        }
        __syncthreads();
        v16h a  = frag_ld_f32(w + (size_t)ot * 16 * DIM + kc * 32, DIM);
        v16h bf = frag_ld_h(&Blds[wv * 16][0], 32);
        acc = wmma_f32_f16(a, bf, acc);
    }
    float* op = outq + ((size_t)(b * 3 * DIM + ot * 16)) * NTOK
                     + tb * 128 + wv * 16;
    frag_st_f32(op, NTOK, acc);
}

// ---------------------------------------------------------------------------
// Kernel 2: depthwise 3x3, SAME padding. Pure memory-bound elementwise.
// ---------------------------------------------------------------------------
__global__ __launch_bounds__(256) void dwconv3x3_kernel(
        const float* __restrict__ in, const float* __restrict__ w9,
        float* __restrict__ out) {
    int idx = blockIdx.x * 256 + threadIdx.x;     // [b][576][4096]
    int s  = idx & (NTOK - 1);
    int ch = (idx >> 12) % (3 * DIM);
    int b  = idx / (3 * DIM * NTOK);
    int y = s >> 6, xx = s & 63;
    const float* ip = in + ((size_t)(b * 3 * DIM + ch)) * NTOK;
    const float* wp = w9 + ch * 9;
    float acc = 0.f;
#pragma unroll
    for (int dy = -1; dy <= 1; ++dy)
#pragma unroll
        for (int dx = -1; dx <= 1; ++dx) {
            int yy = y + dy, xc = xx + dx;
            if (yy >= 0 && yy < HTOK && xc >= 0 && xc < HTOK)
                acc += wp[(dy + 1) * 3 + (dx + 1)] * ip[yy * HTOK + xc];
        }
    out[idx] = acc;
}

// ---------------------------------------------------------------------------
// Kernel 3: split heads, L2-normalize q & k (q pre-scaled by temperature),
// convert to f16.  q,k -> [bh][NTOK][CP] token-major (pad cols zeroed),
// v -> [bh][CP][NTOK] channel-major (so PV B-frags read contiguous rows).
// ---------------------------------------------------------------------------
__global__ __launch_bounds__(256) void build_qkv_kernel(
        const float* __restrict__ qkvdw, const float* __restrict__ temperature,
        _Float16* __restrict__ qb, _Float16* __restrict__ kb,
        _Float16* __restrict__ vtb) {
    const int bh = blockIdx.y;
    const int b = bh >> 2, head = bh & 3;
    const int tok = blockIdx.x * 256 + threadIdx.x;
    const float* base = qkvdw + (size_t)b * 3 * DIM * NTOK + tok;
    const float* qp = base + (size_t)(head * CH) * NTOK;
    const float* kp = base + (size_t)(DIM + head * CH) * NTOK;
    const float* vp = base + (size_t)(2 * DIM + head * CH) * NTOK;

    float qv[CH], kv[CH];
    float qs = 0.f, ks = 0.f;
#pragma unroll
    for (int c = 0; c < CH; ++c) {
        qv[c] = qp[(size_t)c * NTOK]; qs += qv[c] * qv[c];
        kv[c] = kp[(size_t)c * NTOK]; ks += kv[c] * kv[c];
    }
    float t = temperature[head];
    float qsc = t   / fmaxf(sqrtf(qs), 1e-12f);
    float ksc = 1.f / fmaxf(sqrtf(ks), 1e-12f);

    _Float16* qo = qb + ((size_t)bh * NTOK + tok) * CP;
    _Float16* ko = kb + ((size_t)bh * NTOK + tok) * CP;
#pragma unroll
    for (int c = 0; c < CH; ++c) {
        qo[c] = (_Float16)(qv[c] * qsc);
        ko[c] = (_Float16)(kv[c] * ksc);
    }
#pragma unroll
    for (int c = CH; c < CP; ++c) { qo[c] = (_Float16)0.f; ko[c] = (_Float16)0.f; }

    _Float16* vo = vtb + (size_t)bh * CP * NTOK + tok;
#pragma unroll
    for (int c = 0; c < CH; ++c) vo[(size_t)c * NTOK] = (_Float16)vp[(size_t)c * NTOK];
#pragma unroll
    for (int c = CH; c < CP; ++c) vo[(size_t)c * NTOK] = (_Float16)0.f;
}

// ---------------------------------------------------------------------------
// Kernel 4: flash attention. grid=(NTOK/128, B*HEADS), block=256 (8 waves).
// Wave owns a 16-row q strip; the block shares double-buffered K(32x64) /
// Vt(64x32) LDS tiles streamed in with GLOBAL_LOAD_ASYNC_TO_LDS_B128 and
// pipelined on ASYNCcnt, overlapping the next tile's copy with this tile's
// WMMAs + online softmax. P goes through per-wave LDS for D->A layout swap.
// ---------------------------------------------------------------------------
__global__ __launch_bounds__(256) void flash_kernel(
        const _Float16* __restrict__ qb, const _Float16* __restrict__ kb,
        const _Float16* __restrict__ vtb, _Float16* __restrict__ attnout) {
    __shared__ __align__(16) _Float16 Klds[2][32][CP];  // [buf][token][c]
    __shared__ __align__(16) _Float16 Vlds[2][CP][32];  // [buf][c][token]
    __shared__ __align__(16) _Float16 Plds[8][16][32];  // per-wave P tile
    const int qblk = blockIdx.x;
    const int bh   = blockIdx.y;
    const int b = bh >> 2, head = bh & 3;
    const int wv = threadIdx.x >> 5, lane = threadIdx.x & 31;
    const int row0 = qblk * 128 + wv * 16;

    const _Float16* qh = qb  + (size_t)bh * NTOK * CP;
    const _Float16* kh = kb  + (size_t)bh * NTOK * CP;
    const _Float16* vh = vtb + (size_t)bh * CP * NTOK;

    const v16h qa0 = frag_ld_h(qh + (size_t)row0 * CP, CP);
    const v16h qa1 = frag_ld_h(qh + (size_t)row0 * CP + 32, CP);

    // staging coordinates: 16B per thread for each of K and V tiles
    const int tkS = threadIdx.x >> 3, cgS = (threadIdx.x & 7) * 8;  // K
    const int cS  = threadIdx.x >> 2, tgS = (threadIdx.x & 3) * 8;  // V

    float mrow[8], lrow[8];
#pragma unroll
    for (int r = 0; r < 8; ++r) { mrow[r] = -3.0e38f; lrow[r] = 0.f; }
    v8f acc0 = {}, acc1 = {}, acc2 = {};
    const int n  = lane & 15;
    const int mo = (lane >> 4) << 3;

    constexpr int NT = NTOK / 32;

    // prologue: start tile 0 (2 async ops per thread: K slice + V slice)
    cp16_async(kh + (size_t)tkS * CP + cgS, &Klds[0][tkS][cgS]);
    cp16_async(vh + (size_t)cS * NTOK + tgS, &Vlds[0][cS][tgS]);

#pragma unroll 1
    for (int kt = 0; kt < NT; ++kt) {
        const int cur = kt & 1;
        if (kt + 1 < NT) {   // kick off the next tile into the other buffer
            const int tok1 = (kt + 1) * 32;
            cp16_async(kh + (size_t)(tok1 + tkS) * CP + cgS,
                       &Klds[cur ^ 1][tkS][cgS]);
            cp16_async(vh + (size_t)cS * NTOK + tok1 + tgS,
                       &Vlds[cur ^ 1][cS][tgS]);
            wait_async<2>();   // current tile's 2 ops done; next 2 in flight
        } else {
            wait_async<0>();
        }
        __syncthreads();       // whole tile visible to all 8 waves

        // S = q . k^T  (two 16x16 tiles, K-dim 64 = 2 x 32)
        v8f s0 = {}, s1 = {};
        s0 = wmma_f32_f16(qa0, frag_ld_h(&Klds[cur][0][0],  CP), s0);
        s0 = wmma_f32_f16(qa1, frag_ld_h(&Klds[cur][0][32], CP), s0);
        s1 = wmma_f32_f16(qa0, frag_ld_h(&Klds[cur][16][0],  CP), s1);
        s1 = wmma_f32_f16(qa1, frag_ld_h(&Klds[cur][16][32], CP), s1);

        // online softmax, row reductions across the 16-lane halves
#pragma unroll
        for (int r = 0; r < 8; ++r) {
            float v0 = s0[r], v1 = s1[r];
            float mx = fmaxf(v0, v1);
            mx = fmaxf(mx, __shfl_xor(mx, 1, 32));
            mx = fmaxf(mx, __shfl_xor(mx, 2, 32));
            mx = fmaxf(mx, __shfl_xor(mx, 4, 32));
            mx = fmaxf(mx, __shfl_xor(mx, 8, 32));
            float mnew = fmaxf(mrow[r], mx);
            float corr = __expf(mrow[r] - mnew);
            float p0 = __expf(v0 - mnew);
            float p1 = __expf(v1 - mnew);
            float ps = p0 + p1;
            ps += __shfl_xor(ps, 1, 32);
            ps += __shfl_xor(ps, 2, 32);
            ps += __shfl_xor(ps, 4, 32);
            ps += __shfl_xor(ps, 8, 32);
            lrow[r] = lrow[r] * corr + ps;
            mrow[r] = mnew;
            acc0[r] *= corr; acc1[r] *= corr; acc2[r] *= corr;
            Plds[wv][mo + r][n]      = (_Float16)p0;
            Plds[wv][mo + r][n + 16] = (_Float16)p1;
        }

        // O += P(16x32) x V(32x48)  (4th n-tile is the zero pad: skipped)
        v16h pa = frag_ld_h(&Plds[wv][0][0], 32);
        acc0 = wmma_f32_f16(pa, frag_ld_h(&Vlds[cur][0][0],  32), acc0);
        acc1 = wmma_f32_f16(pa, frag_ld_h(&Vlds[cur][16][0], 32), acc1);
        acc2 = wmma_f32_f16(pa, frag_ld_h(&Vlds[cur][32][0], 32), acc2);
        __syncthreads();       // everyone done reading buf `cur` before it
                               // gets re-staged by the next iteration's prefetch
    }

    // epilogue: divide by softmax denominator, emit f16 token-major [b][tok][192]
    _Float16* op = attnout + ((size_t)(b * NTOK + row0)) * DIM + head * CH;
#pragma unroll
    for (int r = 0; r < 8; ++r) {
        float inv = 1.f / lrow[r];
        op[(size_t)(mo + r) * DIM + n]      = (_Float16)(acc0[r] * inv);
        op[(size_t)(mo + r) * DIM + 16 + n] = (_Float16)(acc1[r] * inv);
        op[(size_t)(mo + r) * DIM + 32 + n] = (_Float16)(acc2[r] * inv);
    }
}

// ---------------------------------------------------------------------------
// Kernel 5: proj 1x1 conv == GEMM  Wp[192x192] * Attn[192xNTOK].
// attnout is f16 token-major -> B-fragments load straight from global.
// ---------------------------------------------------------------------------
__global__ __launch_bounds__(256) void proj1x1_kernel(
        const _Float16* __restrict__ attnout, const float* __restrict__ w,
        float* __restrict__ out) {
    const int tb = blockIdx.x;           // 128-token block
    const int ot = blockIdx.y;           // 16-row out-channel tile
    const int b  = blockIdx.z;
    const int wv = threadIdx.x >> 5;
    const int tok0 = tb * 128 + wv * 16;
    v8f acc = {};
#pragma unroll
    for (int kc = 0; kc < DIM / 32; ++kc) {
        v16h a  = frag_ld_f32(w + (size_t)ot * 16 * DIM + kc * 32, DIM);
        v16h bf = frag_ld_h(attnout + ((size_t)(b * NTOK + tok0)) * DIM + kc * 32,
                            DIM);
        acc = wmma_f32_f16(a, bf, acc);
    }
    float* op = out + ((size_t)(b * DIM + ot * 16)) * NTOK + tok0;
    frag_st_f32(op, NTOK, acc);
}

// ---------------------------------------------------------------------------
extern "C" void kernel_launch(void* const* d_in, const int* in_sizes, int n_in,
                              void* d_out, int out_size, void* d_ws, size_t ws_size,
                              hipStream_t stream) {
    const float* x      = (const float*)d_in[0];
    const float* qkv_w  = (const float*)d_in[1];
    const float* dw_w   = (const float*)d_in[2];
    const float* proj_w = (const float*)d_in[3];
    const float* temp   = (const float*)d_in[4];
    float* out = (float*)d_out;

    char*  ws  = (char*)d_ws;
    size_t off = 0;
    auto carve = [&](size_t bytes) -> void* {
        void* p = ws + off;
        off += (bytes + 255) & ~(size_t)255;
        return p;
    };
    float*    qkv    = (float*)   carve((size_t)NB * 3 * DIM * NTOK * 4);
    float*    qkvdw  = (float*)   carve((size_t)NB * 3 * DIM * NTOK * 4);
    _Float16* qbuf   = (_Float16*)carve((size_t)NB * HEADS * NTOK * CP * 2);
    _Float16* kbuf   = (_Float16*)carve((size_t)NB * HEADS * NTOK * CP * 2);
    _Float16* vtbuf  = (_Float16*)carve((size_t)NB * HEADS * CP * NTOK * 2);
    _Float16* aobuf  = (_Float16*)carve((size_t)NB * NTOK * DIM * 2);

    qkv1x1_kernel<<<dim3(NTOK / 128, 3 * DIM / 16, NB), 256, 0, stream>>>(
        x, qkv_w, qkv);
    dwconv3x3_kernel<<<(NB * 3 * DIM * NTOK) / 256, 256, 0, stream>>>(
        qkv, dw_w, qkvdw);
    build_qkv_kernel<<<dim3(NTOK / 256, NB * HEADS), 256, 0, stream>>>(
        qkvdw, temp, qbuf, kbuf, vtbuf);
    flash_kernel<<<dim3(NTOK / 128, NB * HEADS), 256, 0, stream>>>(
        qbuf, kbuf, vtbuf, aobuf);
    proj1x1_kernel<<<dim3(NTOK / 128, DIM / 16, NB), 256, 0, stream>>>(
        aobuf, proj_w, out);
}
